// NeuTraj_Network_19636590477897
// MI455X (gfx1250) — matrix operand order repaired and checked by reading, not compiled, not run on hardware
//
#include <hip/hip_runtime.h>
#include <hip/hip_bf16.h>
#include <math.h>

// ---------------- problem constants (from reference) ----------------
#define WNB   2
#define BB    128
#define TT    200
#define HH    128
#define GRIDC 1100
#define NNG   (GRIDC + 3 * WNB)     // 1106  (memory is NNG x NNG x HH fp32)
#define KNB   25                    // (2W+1)^2 neighborhood
#define FIVEH (5 * HH)              // 640

// LDS strides (elements): 16B-aligned rows + bank spreading
#define WSTR  136                   // w_hh : 640 x 136 bf16 (272B rows)
#define HSTR  136                   // h    : 128 x 136 bf16
#define AQSTR 264                   // [mix|q] : 128 x 264 bf16 (256 cols used)

typedef __attribute__((ext_vector_type(16))) __bf16 v16bf;
typedef __attribute__((ext_vector_type(8)))  float  v8f;

union FragAB {
    v16bf    v;
    uint4    q[2];
    unsigned u[8];
};

// ---------------- scalar helpers ----------------
__device__ __forceinline__ unsigned short f2bf(float f) {
    unsigned u = __float_as_uint(f);
    unsigned r = (u + 0x7FFFu + ((u >> 16) & 1u)) >> 16;   // RNE
    return (unsigned short)r;
}
__device__ __forceinline__ float bf2f(unsigned short h) {
    return __uint_as_float(((unsigned)h) << 16);
}
__device__ __forceinline__ float sigf(float x) {
    return __builtin_amdgcn_rcpf(1.0f + __expf(-x));       // v_exp + v_rcp
}
__device__ __forceinline__ float tanhf_fast(float x) {
    return 1.0f - 2.0f * __builtin_amdgcn_rcpf(1.0f + __expf(2.0f * x));
}
__device__ __forceinline__ float wave_sum(float v) {
    #pragma unroll
    for (int o = 16; o; o >>= 1) v += __shfl_xor(v, o, 32);
    return v;
}
__device__ __forceinline__ float wave_max(float v) {
    #pragma unroll
    for (int o = 16; o; o >>= 1) v = fmaxf(v, __shfl_xor(v, o, 32));
    return v;
}

// ---------------- WMMA fragment loaders (contiguous K order, same for A and B) ----------------
// A 16x32 bf16 (MxK): lane l: M = mi*16 + l%16; lanes<16 hold K {0..15},
// lanes>=16 hold K {16..31} => two ds_load_b128 per fragment.
__device__ __forceinline__ void load_fragA_lds(FragAB& f, const unsigned short* base,
                                               int mi, int kb, int stride, int lane) {
    const int half = lane >> 4;
    const int m    = mi * 16 + (lane & 15);
    const unsigned short* p = base + m * stride + kb * 32 + half * 16;
    f.q[0] = *reinterpret_cast<const uint4*>(p);        // K k0..k0+7
    f.q[1] = *reinterpret_cast<const uint4*>(p + 8);    // K k0+8..k0+15
}
// B 32x16 bf16 (KxN): lane l: N = nbase + l%16; K contiguous per lane half.
// B(k,n) = w[n*ld + k] (weights row-major (out,in)) => transpose-free.
__device__ __forceinline__ void load_fragB_lds(FragAB& f, const unsigned short* w,
                                               int nbase, int kb, int lane) {
    const int half = lane >> 4;
    const int n    = nbase + (lane & 15);
    const unsigned short* p = w + n * WSTR + kb * 32 + half * 16;
    f.q[0] = *reinterpret_cast<const uint4*>(p);        // K k0..k0+7
    f.q[1] = *reinterpret_cast<const uint4*>(p + 8);    // K k0+8..k0+15
}
__device__ __forceinline__ void load_fragB_glb(FragAB& f, const unsigned short* w, int ld,
                                               int nbase, int kb, int lane) {
    const int half = lane >> 4;
    const int n    = nbase + (lane & 15);
    const unsigned short* p = w + (size_t)n * ld + kb * 32 + half * 16;
    f.q[0] = *reinterpret_cast<const uint4*>(p);
    f.q[1] = *reinterpret_cast<const uint4*>(p + 8);
}

// -------- LDS layout (dynamic shared, ~296.5 KB of the WGP's 320 KB) --------
#define OFF_WHH    0                                  // 640*136 u16 = 174080
#define OFF_H      174080                             // 128*136 u16 = 34816
#define OFF_AQ     208896                             // 128*264 u16 = 67584
#define OFF_SC     276480                             // 128*32  f32 = 16384
#define OFF_F0     292864                             // 128 f32
#define OFF_F1     293376
#define OFF_GX     293888                             // 128 i32
#define OFF_GY     294400
#define OFF_BIAS   294912                             // 640 f32 (b_ih+b_hh)
#define OFF_WIH    297472                             // 640*2 f32
#define OFF_ATB    302592                             // 128 f32
#define OFF_LEN    303104                             // 128 i32
#define SMEM_BYTES 303616

// ---------------- prep kernels ----------------
__global__ void prep_zero_f4(float4* __restrict__ p, long n4) {
    long i      = (long)blockIdx.x * blockDim.x + threadIdx.x;
    long stride = (long)gridDim.x * blockDim.x;
    float4 z = make_float4(0.f, 0.f, 0.f, 0.f);
    for (; i < n4; i += stride) p[i] = z;
}
__global__ void prep_cvt_bf16(const float* __restrict__ s, unsigned short* __restrict__ d, int n) {
    int i = blockIdx.x * blockDim.x + threadIdx.x;
    if (i < n) d[i] = f2bf(s[i]);
}

// ---------------- persistent single-WGP recurrent kernel ----------------
__global__ void __launch_bounds__(512, 1)
neutraj_persist(const float* __restrict__ inp,     // (B,T,4)
                const int*   __restrict__ lens,    // (B,)
                const float* __restrict__ w_ih,    // (640,2)
                const float* __restrict__ w_hh,    // (640,128) fp32
                const float* __restrict__ b_ih,    // (640,)
                const float* __restrict__ b_hh,    // (640,)
                const float* __restrict__ attn_b,  // (128,)
                const float* __restrict__ h0,      // (B,H)
                const float* __restrict__ c0,      // (B,H)
                float*       __restrict__ memF,    // ws: (N,N,H) fp32, pre-zeroed
                const unsigned short* __restrict__ attnWb, // ws: (128,256) bf16
                float*       __restrict__ out)     // (B,H)
{
    extern __shared__ __align__(16) unsigned char smem[];
    unsigned short* wL   = (unsigned short*)(smem + OFF_WHH);  // gates B operand
    unsigned short* hL   = (unsigned short*)(smem + OFF_H);
    unsigned short* aq   = (unsigned short*)(smem + OFF_AQ);   // [0,128)=mix, [128,256)=q(=cy)
    float*          scL  = (float*)         (smem + OFF_SC);
    float*          f0L  = (float*)         (smem + OFF_F0);
    float*          f1L  = (float*)         (smem + OFF_F1);
    int*            gXi  = (int*)           (smem + OFF_GX);
    int*            gYi  = (int*)           (smem + OFF_GY);
    float*          biasC= (float*)         (smem + OFF_BIAS);
    float*          wihL = (float*)         (smem + OFF_WIH);
    float*          atbL = (float*)         (smem + OFF_ATB);
    int*            lenL = (int*)           (smem + OFF_LEN);

    const int tid  = threadIdx.x;
    const int lane = tid & 31;
    const int wave = tid >> 5;          // 0..15
    const int half = lane >> 4;

    // ---- one-time init ----
    for (int i = tid; i < FIVEH; i += 512) {
        biasC[i]      = b_ih[i] + b_hh[i];
        wihL[2*i]     = w_ih[2*i];
        wihL[2*i + 1] = w_ih[2*i + 1];
    }
    if (tid < HH) { atbL[tid] = attn_b[tid]; lenL[tid] = lens[tid]; }
    for (int i = tid; i < BB * HH; i += 512) {            // h0 -> LDS (bf16)
        int r = i >> 7, c = i & 127;
        hL[r * HSTR + c] = f2bf(h0[i]);
    }
    for (int i = tid; i < FIVEH * HH; i += 512) {         // w_hh -> LDS (bf16, padded)
        int r = i >> 7, c = i & 127;
        wL[r * WSTR + c] = f2bf(w_hh[i]);
    }

    // persistent per-wave state: each wave owns 4 (ti,tj) tiles; lane layout
    // matches the WMMA C/D 16x16 f32 fragment (row = ti*16 + half*8 + r).
    v8f cC[4], sgR[4], ogR[4];
    #pragma unroll
    for (int pp = 0; pp < 4; ++pp) {
        const int p  = wave * 4 + pp;
        const int ti = p >> 3, tj = p & 7;
        const int col = tj * 16 + (lane & 15);
        #pragma unroll
        for (int r = 0; r < 8; ++r) {
            const int row = ti * 16 + half * 8 + r;
            cC[pp][r] = c0[row * HH + col];
            sgR[pp][r] = 0.0f; ogR[pp][r] = 0.0f;
        }
    }
    __syncthreads();

    for (int t = 0; t < TT; ++t) {
        // ============ phase 1: stage x_t ============
        if (tid < BB) {
            const float* x = inp + ((size_t)tid * TT + t) * 4;
            f0L[tid] = x[0];
            f1L[tid] = x[1];
            gXi[tid] = (int)x[2] + WNB;
            gYi[tid] = (int)x[3] + WNB;
        }
        __syncthreads();

        // ============ phase 2: gates GEMM (128 x 640, K=128), B from LDS ====
        #pragma unroll
        for (int pp = 0; pp < 4; ++pp) {
            const int p  = wave * 4 + pp;
            const int ti = p >> 3, tj = p & 7;
            FragAB A[4];
            #pragma unroll
            for (int kb = 0; kb < 4; ++kb) load_fragA_lds(A[kb], hL, ti, kb, HSTR, lane);

            const int colc = tj * 16 + (lane & 15);     // column within a chunk
            v8f acc[5];
            #pragma unroll
            for (int g = 0; g < 5; ++g) {
                const int col = g * 128 + colc;
                const float bias = biasC[col], w0 = wihL[2*col], w1 = wihL[2*col + 1];
                #pragma unroll
                for (int r = 0; r < 8; ++r) {
                    const int row = ti * 16 + half * 8 + r;
                    acc[g][r] = bias + f0L[row] * w0 + f1L[row] * w1;
                }
            }
            #pragma unroll
            for (int g = 0; g < 5; ++g) {
                #pragma unroll
                for (int kb = 0; kb < 4; ++kb) {
                    FragAB Bf;
                    load_fragB_lds(Bf, wL, g * 128 + tj * 16, kb, lane);
                    acc[g] = __builtin_amdgcn_wmma_f32_16x16x32_bf16(
                        false, A[kb].v, false, Bf.v, (short)0, acc[g], false, false);
                }
            }
            // fused activations + LSTM cell pre-attention (state in registers)
            #pragma unroll
            for (int r = 0; r < 8; ++r) {
                const int row = ti * 16 + half * 8 + r;
                const float ig = sigf(acc[0][r]);
                const float fg = sigf(acc[1][r]);
                const float gg = tanhf_fast(acc[2][r]);
                ogR[pp][r] = sigf(acc[3][r]);
                sgR[pp][r] = sigf(acc[4][r]);
                const float cy = fg * cC[pp][r] + ig * gg;
                aq[row * AQSTR + HH + colc] = f2bf(cy);   // q half of attn input
            }
        }
        __syncthreads();

        // ============ phase 4: spatial attention gather + softmax + mix ====
        for (int bi = 0; bi < 8; ++bi) {
            const int b  = wave * 8 + bi;
            const int gx = gXi[b], gy = gYi[b];
            float4 q4;
            {
                const uint2 uu = *reinterpret_cast<const uint2*>(aq + b * AQSTR + HH + lane * 4);
                q4.x = bf2f((unsigned short)(uu.x & 0xffff));
                q4.y = bf2f((unsigned short)(uu.x >> 16));
                q4.z = bf2f((unsigned short)(uu.y & 0xffff));
                q4.w = bf2f((unsigned short)(uu.y >> 16));
            }
            #pragma unroll
            for (int k = 0; k < KNB; ++k) {
                const int dx = k / 5 - WNB, dy = k % 5 - WNB;
                const float4 cs = *reinterpret_cast<const float4*>(
                    memF + ((size_t)(gx + dx) * NNG + (size_t)(gy + dy)) * HH + lane * 4);
                float s = q4.x * cs.x + q4.y * cs.y + q4.z * cs.z + q4.w * cs.w;
                s = wave_sum(s);
                if (lane == 0) scL[b * 32 + k] = s;
            }
            {
                const float sc  = (lane < KNB) ? scL[b * 32 + lane] : 0.0f;
                const float msk = (sc == 0.0f) ? -__builtin_inff() : sc;
                const float mx  = wave_max(msk);
                const float e   = (sc == 0.0f) ? 0.0f : __expf(sc - mx);
                const float sm  = wave_sum(e);
                const float a   = (sm > 0.0f) ? e * __builtin_amdgcn_rcpf(sm) : 0.0f;
                if (lane < KNB) scL[b * 32 + lane] = a;
            }
            float4 mix = make_float4(0.f, 0.f, 0.f, 0.f);
            #pragma unroll
            for (int k = 0; k < KNB; ++k) {
                const float ak = scL[b * 32 + k];
                const int dx = k / 5 - WNB, dy = k % 5 - WNB;
                const float4 cs = *reinterpret_cast<const float4*>(
                    memF + ((size_t)(gx + dx) * NNG + (size_t)(gy + dy)) * HH + lane * 4);
                mix.x += ak * cs.x; mix.y += ak * cs.y; mix.z += ak * cs.z; mix.w += ak * cs.w;
            }
            uint2 o;
            o.x = (unsigned)f2bf(mix.x) | ((unsigned)f2bf(mix.y) << 16);
            o.y = (unsigned)f2bf(mix.z) | ((unsigned)f2bf(mix.w) << 16);
            *reinterpret_cast<uint2*>(aq + b * AQSTR + lane * 4) = o;   // mix half
        }
        __syncthreads();

        // ============ phase 5: attn GEMM (128 x 128, K=256) + finalize =====
        #pragma unroll
        for (int pp = 0; pp < 4; ++pp) {
            const int p  = wave * 4 + pp;
            const int ti = p >> 3, tj = p & 7;
            const int col = tj * 16 + (lane & 15);
            // prefetch all 8 L2-resident B fragments (attn_w is step-invariant)
            FragAB Bfs[8];
            #pragma unroll
            for (int kb = 0; kb < 8; ++kb)
                load_fragB_glb(Bfs[kb], attnWb, 2 * HH, tj * 16, kb, lane);
            v8f acc;
            #pragma unroll
            for (int r = 0; r < 8; ++r) acc[r] = atbL[col];
            #pragma unroll
            for (int kb = 0; kb < 8; ++kb) {
                FragAB Aa;
                load_fragA_lds(Aa, aq, ti, kb, AQSTR, lane);
                acc = __builtin_amdgcn_wmma_f32_16x16x32_bf16(
                    false, Aa.v, false, Bfs[kb].v, (short)0, acc, false, false);
            }
            #pragma unroll
            for (int r = 0; r < 8; ++r) {
                const int row = ti * 16 + half * 8 + r;
                const float att = tanhf_fast(acc[r]);
                const float cy  = bf2f(aq[row * AQSTR + HH + col]);
                const float sg  = sgR[pp][r];
                const float cn  = cy + sg * att;
                cC[pp][r] = cn;                                // fp32 cell state
                const float hy = ogR[pp][r] * tanhf_fast(cn);
                hL[row * HSTR + col] = f2bf(hy);               // next-step A operand
                float* mp = memF + ((size_t)gXi[row] * NNG + (size_t)gYi[row]) * HH + col;
                const float oldv = *mp;
                *mp = sg * oldv + (1.0f - sg) * cn;            // sg*old + (1-sg)*c_new
                if (t == lenL[row] - 1) out[row * HH + col] = hy;
            }
        }
        __threadfence();        // memory writes visible to next step's gathers
        __syncthreads();
    }
}

// ---------------- host entry ----------------
extern "C" void kernel_launch(void* const* d_in, const int* in_sizes, int n_in,
                              void* d_out, int out_size, void* d_ws, size_t ws_size,
                              hipStream_t stream) {
    const float* inp    = (const float*)d_in[0];
    const int*   lens   = (const int*)  d_in[1];
    const float* w_ih   = (const float*)d_in[2];
    const float* w_hh   = (const float*)d_in[3];
    const float* b_ih   = (const float*)d_in[4];
    const float* b_hh   = (const float*)d_in[5];
    const float* attn_w = (const float*)d_in[6];
    const float* attn_b = (const float*)d_in[7];
    const float* h0     = (const float*)d_in[8];
    const float* c0     = (const float*)d_in[9];
    // d_in[10] = memory (all zeros): inputs are immutable, so the mutable
    // memory tensor lives in d_ws and is re-zeroed every call (deterministic).
    float* out = (float*)d_out;

    const size_t memElems = (size_t)NNG * NNG * HH;            // 156,574,208 f32
    const size_t memBytes = memElems * sizeof(float);          // ~626.3 MB
    float*          memF   = (float*)d_ws;
    unsigned short* attnWb = (unsigned short*)((char*)d_ws + memBytes);

    prep_zero_f4<<<4096, 256, 0, stream>>>((float4*)memF, (long)(memElems / 4));
    prep_cvt_bf16<<<(HH * 2 * HH + 255) / 256, 256, 0, stream>>>(attn_w, attnWb, HH * 2 * HH);

    (void)hipFuncSetAttribute((const void*)neutraj_persist,
                              hipFuncAttributeMaxDynamicSharedMemorySize, SMEM_BYTES);
    neutraj_persist<<<1, 512, SMEM_BYTES, stream>>>(
        inp, lens, w_ih, w_hh, b_ih, b_hh, attn_b, h0, c0, memF, attnWb, out);
}
